// TensoRF_79748952752847
// MI455X (gfx1250) — matrix operand order repaired and checked by reading, not compiled, not run on hardware
//
#include <hip/hip_runtime.h>

#define GSZ   128
#define GSZ2  (128*128)
#define GSZ3  (128*128*128)
#define NSAMP 128
#define NRAYS 32768

typedef __attribute__((ext_vector_type(16))) _Float16 v16h;
typedef __attribute__((ext_vector_type(8)))  float    v8f;

__device__ __forceinline__ int iclamp(int v, int lo, int hi) {
    return v < lo ? lo : (v > hi ? hi : v);
}

// Wave-local LDS fence: LDS ops from one wave complete in order (CDNA5 ISA
// 7.3), so for per-wave LDS tiles a compiler fence + s_wait_dscnt 0 replaces
// a full workgroup barrier.
#define WAVE_LDS_SYNC() asm volatile("s_wait_dscnt 0x0" ::: "memory")

// One wave per ray.  8 waves per block.  All LDS is partitioned per-wave, so
// no workgroup barriers are needed anywhere.
__global__ __launch_bounds__(256) void nerf_render_kernel(
    const float* __restrict__ rays,
    const float* __restrict__ grid,   // (32,128,128,128)
    const float* __restrict__ dw1,    // (32,64)
    const float* __restrict__ db1,    // (64)
    const float* __restrict__ dw2,    // (64,1)
    const float* __restrict__ db2,    // (1)
    const float* __restrict__ cw1,    // (35,64)
    const float* __restrict__ cb1,    // (64)
    const float* __restrict__ cw2,    // (64,3)
    const float* __restrict__ cb2,    // (3)
    float* __restrict__ out)          // (32768,3)
{
    __shared__ _Float16 feat[8][16 * 34];      // 16 samples x 32 ch  (stride 34 halves)
    __shared__ _Float16 h16 [8][16 * 68];      // 16 samples x 64 hid (stride 68 halves)
    __shared__ float    dens[8][NSAMP];
    __shared__ float    cols[8][NSAMP * 3];

    const int wave = threadIdx.x >> 5;         // 0..7
    const int lane = threadIdx.x & 31;
    const int half = lane >> 4;                // 0 or 1
    const int ln   = lane & 15;
    const int ray  = blockIdx.x * 8 + wave;
    const int mbase = half * 8;

    // --- ray (uniform across the wave) ---
    const float ox = rays[ray * 6 + 0], oy = rays[ray * 6 + 1], oz = rays[ray * 6 + 2];
    const float dx = rays[ray * 6 + 3], dy = rays[ray * 6 + 4], dz = rays[ray * 6 + 5];

    // --- layer-1 B fragments (f16, ISA 32x16 layout: k = (j<8 ? j : j+8) + 8*half, n = ln) ---
    v16h b_dens[4], b_col[4];
    #pragma unroll
    for (int t4 = 0; t4 < 4; ++t4) {
        #pragma unroll
        for (int j = 0; j < 16; ++j) {
            const int k = (j < 8 ? j : j + 8) + half * 8;
            const int n = t4 * 16 + ln;
            b_dens[t4][j] = (_Float16)dw1[k * 64 + n];
            b_col[t4][j]  = (_Float16)cw1[k * 64 + n];
        }
    }
    // --- layer-2 B fragments: 64x16, zero-padded columns ---
    // density: column 0 = dw2 ; color: columns 0..2 = cw2
    v16h b2d[2], b2c[2];
    #pragma unroll
    for (int f = 0; f < 2; ++f) {
        #pragma unroll
        for (int j = 0; j < 16; ++j) {
            const int k = f * 32 + (j < 8 ? j : j + 8) + half * 8;       // K = 0..63
            b2d[f][j] = (ln == 0) ? (_Float16)dw2[k]          : (_Float16)0.0f;
            b2c[f][j] = (ln <  3) ? (_Float16)cw2[k * 3 + ln] : (_Float16)0.0f;
        }
    }

    // --- C-matrix inits ---
    float cinit_d[4], cinit_c[4];
    #pragma unroll
    for (int t4 = 0; t4 < 4; ++t4) {
        const int n = t4 * 16 + ln;
        cinit_d[t4] = db1[n];
        cinit_c[t4] = cb1[n] + dx * cw1[32 * 64 + n]
                             + dy * cw1[33 * 64 + n]
                             + dz * cw1[34 * 64 + n];
    }
    const float cinit2_d = (ln == 0) ? db2[0]  : 0.0f;
    const float cinit2_c = (ln <  3) ? cb2[ln] : 0.0f;

    for (int tile = 0; tile < 8; ++tile) {
        // ---------- trilerp: sample = ln, channels half*16..+15 ----------
        {
            const int   s  = tile * 16 + ln;
            const float tv = (float)s * (1.0f / 127.0f);
            const float px = ox + tv * dx, py = oy + tv * dy, pz = oz + tv * dz;
            const float cx = (px + 1.0f) * 0.5f * 127.0f;
            const float cy = (py + 1.0f) * 0.5f * 127.0f;
            const float cz = (pz + 1.0f) * 0.5f * 127.0f;
            const int x0 = iclamp((int)floorf(cx), 0, GSZ - 1);
            const int y0 = iclamp((int)floorf(cy), 0, GSZ - 1);
            const int z0 = iclamp((int)floorf(cz), 0, GSZ - 1);
            const int x1 = x0 + 1 > GSZ - 1 ? GSZ - 1 : x0 + 1;
            const int y1 = y0 + 1 > GSZ - 1 ? GSZ - 1 : y0 + 1;
            const int z1 = z0 + 1 > GSZ - 1 ? GSZ - 1 : z0 + 1;
            const float xd = cx - (float)x0, yd = cy - (float)y0, zd = cz - (float)z0;
            const float xi = 1.0f - xd, yi = 1.0f - yd, zi = 1.0f - zd;
            const int o000 = z0 * GSZ2 + y0 * GSZ + x0;
            const int o001 = z0 * GSZ2 + y0 * GSZ + x1;
            const int o010 = z0 * GSZ2 + y1 * GSZ + x0;
            const int o100 = z1 * GSZ2 + y0 * GSZ + x0;
            const int o011 = z0 * GSZ2 + y1 * GSZ + x1;
            const int o101 = z1 * GSZ2 + y0 * GSZ + x1;
            const int o110 = z1 * GSZ2 + y1 * GSZ + x0;
            const int o111 = z1 * GSZ2 + y1 * GSZ + x1;
            const float w000 = xi * yi * zi, w001 = xd * yi * zi;
            const float w010 = xi * yd * zi, w100 = xi * yi * zd;
            const float w011 = xd * yd * zi, w101 = xd * yi * zd;
            const float w110 = xi * yd * zd, w111 = xd * yd * zd;
            for (int i = 0; i < 16; ++i) {
                const int c = half * 16 + i;
                const float* gc = grid + (size_t)c * GSZ3;
                float v = w000 * gc[o000] + w001 * gc[o001]
                        + w010 * gc[o010] + w100 * gc[o100]
                        + w011 * gc[o011] + w101 * gc[o101]
                        + w110 * gc[o110] + w111 * gc[o111];
                feat[wave][ln * 34 + c] = (_Float16)v;
            }
        }
        WAVE_LDS_SYNC();

        // ---------- A fragment (16x32 f16, ISA layout) ----------
        v16h a;
        #pragma unroll
        for (int j = 0; j < 16; ++j) {
            const int k = (j < 8 ? j : j + 8) + half * 8;
            a[j] = feat[wave][ln * 34 + k];
        }

        // ================= DENSITY branch =================
        // layer 1: 4 x v_wmma_f32_16x16x32_f16, relu -> f16 -> LDS (A-layout)
        #pragma unroll
        for (int t4 = 0; t4 < 4; ++t4) {
            v8f c8;
            #pragma unroll
            for (int r = 0; r < 8; ++r) c8[r] = cinit_d[t4];
            c8 = __builtin_amdgcn_wmma_f32_16x16x32_f16(
                     false, a, false, b_dens[t4], (short)0, c8, false, false);
            const int n = t4 * 16 + ln;
            #pragma unroll
            for (int r = 0; r < 8; ++r)
                h16[wave][(mbase + r) * 68 + n] = (_Float16)fmaxf(c8[r], 0.0f);
        }
        WAVE_LDS_SYNC();

        // layer 2 (64 -> 1 via zero-padded 64x16): K=64 -> 2 chained WMMAs
        {
            v16h a20, a21;
            #pragma unroll
            for (int j = 0; j < 16; ++j) {
                const int k = (j < 8 ? j : j + 8) + half * 8;
                a20[j] = h16[wave][ln * 68 + k];
                a21[j] = h16[wave][ln * 68 + 32 + k];
            }
            v8f c8;
            #pragma unroll
            for (int r = 0; r < 8; ++r) c8[r] = cinit2_d;
            c8 = __builtin_amdgcn_wmma_f32_16x16x32_f16(
                     false, a20, false, b2d[0], (short)0, c8, false, false);
            c8 = __builtin_amdgcn_wmma_f32_16x16x32_f16(
                     false, a21, false, b2d[1], (short)0, c8, false, false);
            if (ln == 0) {
                #pragma unroll
                for (int r = 0; r < 8; ++r)
                    dens[wave][tile * 16 + mbase + r] = c8[r];
            }
        }
        WAVE_LDS_SYNC();   // h16 WAR before the color branch overwrites it

        // ================= COLOR branch =================
        // layer 1: 4 x v_wmma_f32_16x16x32_f16 (direction term in C-init)
        #pragma unroll
        for (int t4 = 0; t4 < 4; ++t4) {
            v8f c8;
            #pragma unroll
            for (int r = 0; r < 8; ++r) c8[r] = cinit_c[t4];
            c8 = __builtin_amdgcn_wmma_f32_16x16x32_f16(
                     false, a, false, b_col[t4], (short)0, c8, false, false);
            const int n = t4 * 16 + ln;
            #pragma unroll
            for (int r = 0; r < 8; ++r)
                h16[wave][(mbase + r) * 68 + n] = (_Float16)fmaxf(c8[r], 0.0f);
        }
        WAVE_LDS_SYNC();

        // layer 2 (64 -> 3 via zero-padded 64x16): K=64 -> 2 chained WMMAs
        {
            v16h a20, a21;
            #pragma unroll
            for (int j = 0; j < 16; ++j) {
                const int k = (j < 8 ? j : j + 8) + half * 8;
                a20[j] = h16[wave][ln * 68 + k];
                a21[j] = h16[wave][ln * 68 + 32 + k];
            }
            v8f c8;
            #pragma unroll
            for (int r = 0; r < 8; ++r) c8[r] = cinit2_c;
            c8 = __builtin_amdgcn_wmma_f32_16x16x32_f16(
                     false, a20, false, b2c[0], (short)0, c8, false, false);
            c8 = __builtin_amdgcn_wmma_f32_16x16x32_f16(
                     false, a21, false, b2c[1], (short)0, c8, false, false);
            if (ln < 3) {
                #pragma unroll
                for (int r = 0; r < 8; ++r)
                    cols[wave][(tile * 16 + mbase + r) * 3 + ln] = c8[r];
            }
        }
        WAVE_LDS_SYNC();
    }

    // ---------- volume rendering integral (serial scan, 3 lanes) ----------
    if (lane < 3) {
        float T = 1.0f, acc = 0.0f;
        const float dt = 1.0f / 127.0f;
        for (int i = 0; i < NSAMP; ++i) {
            const float delta = (i == NSAMP - 1) ? 1e10f : dt;
            const float alpha = 1.0f - expf(-dens[wave][i] * delta);
            acc += alpha * T * cols[wave][i * 3 + lane];
            T *= (1.0f - alpha + 1e-10f);
        }
        out[ray * 3 + lane] = acc;
    }
}

extern "C" void kernel_launch(void* const* d_in, const int* in_sizes, int n_in,
                              void* d_out, int out_size, void* d_ws, size_t ws_size,
                              hipStream_t stream) {
    const float* rays = (const float*)d_in[0];
    const float* grid = (const float*)d_in[1];
    const float* dw1  = (const float*)d_in[2];
    const float* db1  = (const float*)d_in[3];
    const float* dw2  = (const float*)d_in[4];
    const float* db2  = (const float*)d_in[5];
    const float* cw1  = (const float*)d_in[6];
    const float* cb1  = (const float*)d_in[7];
    const float* cw2  = (const float*)d_in[8];
    const float* cb2  = (const float*)d_in[9];
    float* out = (float*)d_out;

    dim3 gdim(NRAYS / 8);
    dim3 bdim(256);
    nerf_render_kernel<<<gdim, bdim, 0, stream>>>(
        rays, grid, dw1, db1, dw2, db2, cw1, cb1, cw2, cb2, out);
}
